// SageModel_41257455845904
// MI455X (gfx1250) — matrix operand order repaired and checked by reading, not compiled, not run on hardware
//
#include <hip/hip_runtime.h>
#include <hip/hip_bf16.h>

typedef __attribute__((ext_vector_type(16))) _Float16 v16h;
typedef __attribute__((ext_vector_type(8)))  _Float16 v8h;
typedef __attribute__((ext_vector_type(2)))  _Float16 v2h;
typedef __attribute__((ext_vector_type(8)))  float    v8f;
typedef __attribute__((ext_vector_type(4)))  float    v4f;

#define FEAT        768
#define LAYER       128
#define LDS_STRIDE  776          // 768 + 8 halfs pad -> row stride 388 dwords == 4 (mod 64 banks)
#define WAVES       8
#define NUM_WG      256
#define LDS_BYTES   (LAYER * LDS_STRIDE * 2)

__device__ __forceinline__ float fast_tanh(float x) {
#if __has_builtin(__builtin_amdgcn_tanhf)
  return __builtin_amdgcn_tanhf(x);      // v_tanh_f32 on gfx1250
#else
  return tanhf(x);
#endif
}

__device__ __forceinline__ float fast_rcp(float x) {
#if __has_builtin(__builtin_amdgcn_rcpf)
  return __builtin_amdgcn_rcpf(x);
#else
  return 1.0f / x;
#endif
}

__global__ __launch_bounds__(256) void sage_fused_kernel(
    const float* __restrict__ X,
    const float* __restrict__ W_R,
    const float* __restrict__ b_R,
    const float* __restrict__ w_A,
    const float* __restrict__ W_D,
    const float* __restrict__ b_D,
    float* __restrict__ out,
    int nTiles)
{
  extern __shared__ _Float16 ldsW[];   // [LAYER][LDS_STRIDE] f16
  const int tid  = threadIdx.x;
  const int lane = tid & 31;
  const int wave = tid >> 5;
  const int cn   = lane & 15;          // column-in-tile / row-in-tile selector
  const int h    = lane >> 4;          // half-wave id

  // ---- Stage W_R: f32 global -> f16 LDS (padded rows), coalesced float2 reads ----
  const float2* WR2 = (const float2*)W_R;
  for (int p = tid; p < (LAYER * FEAT) / 2; p += 256) {
    int e   = p * 2;
    int row = e / FEAT;
    int col = e - row * FEAT;
    float2 f = WR2[p];
    v2h hp; hp[0] = (_Float16)f.x; hp[1] = (_Float16)f.y;
    *(v2h*)(ldsW + row * LDS_STRIDE + col) = hp;
  }
  __syncthreads();

  // ---- Per-lane epilogue constants: n = 16*t + cn for t = 0..7 ----
  float br[8], wa[8], wd0[8], wd1[8];
#pragma unroll
  for (int t = 0; t < 8; ++t) {
    int n = t * 16 + cn;
    br[t]  = b_R[n];
    wa[t]  = w_A[n];
    wd0[t] = W_D[n];            // W_D[0][n]
    wd1[t] = W_D[LAYER + n];    // W_D[1][n]
  }
  const float bd0 = b_D[0], bd1 = b_D[1];

  for (int tileIdx = blockIdx.x * WAVES + wave; tileIdx < nTiles;
       tileIdx += gridDim.x * WAVES) {
    const int row0 = tileIdx * 16;                 // 16 rows = 8 samples x 2 branches
    const float* arow = X + (size_t)(row0 + cn) * FEAT + h * 8;

    v8f acc[8];
#pragma unroll
    for (int t = 0; t < 8; ++t) acc[t] = (v8f)0.0f;

    // ---- GEMM: K loop, 32 per step; A from global (f32->f16), B from LDS ----
    for (int kb = 0; kb < FEAT; kb += 32) {
      v4f f0 = __builtin_nontemporal_load((const v4f*)(arow + kb));
      v4f f1 = __builtin_nontemporal_load((const v4f*)(arow + kb + 4));
      v4f f2 = __builtin_nontemporal_load((const v4f*)(arow + kb + 16));
      v4f f3 = __builtin_nontemporal_load((const v4f*)(arow + kb + 20));
      v16h a;
#pragma unroll
      for (int i = 0; i < 4; ++i) {
        a[i]      = (_Float16)f0[i];   // K = kb + 8h + i
        a[4 + i]  = (_Float16)f1[i];   // K = kb + 8h + 4 + i
        a[8 + i]  = (_Float16)f2[i];   // K = kb + 16 + 8h + i
        a[12 + i] = (_Float16)f3[i];   // K = kb + 16 + 8h + 4 + i
      }
      const _Float16* bp = ldsW + (size_t)cn * LDS_STRIDE + kb + h * 8;
#pragma unroll
      for (int t = 0; t < 8; ++t) {
        const _Float16* q = bp + t * 16 * LDS_STRIDE;   // row n = 16t + cn
        v8h blo = *(const v8h*)(q);                      // K = kb+8h .. +7
        v8h bhi = *(const v8h*)(q + 16);                 // K = kb+16+8h .. +7
        v16h b = __builtin_shufflevector(blo, bhi, 0, 1, 2, 3, 4, 5, 6, 7,
                                         8, 9, 10, 11, 12, 13, 14, 15);
        acc[t] = __builtin_amdgcn_wmma_f32_16x16x32_f16(
            false, a, false, b, (short)0, acc[t], false, false);
      }
    }

    // ---- Epilogue: bias + leaky relu; attention logits per row ----
    // C layout: acc[t][j] at lane L holds r[row = j + 8h][n = 16t + cn]
    float att[8];
#pragma unroll
    for (int j = 0; j < 8; ++j) att[j] = 0.0f;
#pragma unroll
    for (int t = 0; t < 8; ++t) {
#pragma unroll
      for (int j = 0; j < 8; ++j) {
        float v = acc[t][j] + br[t];
        v = (v >= 0.0f) ? v : 0.01f * v;
        acc[t][j] = v;
        att[j] += fast_tanh(v) * wa[t];
      }
    }
#pragma unroll
    for (int j = 0; j < 8; ++j) {          // reduce over the 16 lanes of each half
      att[j] += __shfl_xor(att[j], 1, 32);
      att[j] += __shfl_xor(att[j], 2, 32);
      att[j] += __shfl_xor(att[j], 4, 32);
      att[j] += __shfl_xor(att[j], 8, 32);
    }

    // ---- 2-way softmax per sample (rows 2i, 2i+1), weighted sum, relu, dense ----
    float q0[4], q1[4];
#pragma unroll
    for (int i = 0; i < 4; ++i) {
      float m  = fmaxf(att[2 * i], att[2 * i + 1]);
      float e0 = __expf(att[2 * i] - m);
      float e1 = __expf(att[2 * i + 1] - m);
      float inv = fast_rcp(e0 + e1);
      att[2 * i]     = e0 * inv;   // a0
      att[2 * i + 1] = e1 * inv;   // a1
      q0[i] = 0.0f; q1[i] = 0.0f;
    }
#pragma unroll
    for (int t = 0; t < 8; ++t) {
#pragma unroll
      for (int i = 0; i < 4; ++i) {
        float s = att[2 * i] * acc[t][2 * i] + att[2 * i + 1] * acc[t][2 * i + 1];
        s = fmaxf(s, 0.0f);
        q0[i] += s * wd0[t];
        q1[i] += s * wd1[t];
      }
    }
#pragma unroll
    for (int i = 0; i < 4; ++i) {
      q0[i] += __shfl_xor(q0[i], 1, 32);
      q0[i] += __shfl_xor(q0[i], 2, 32);
      q0[i] += __shfl_xor(q0[i], 4, 32);
      q0[i] += __shfl_xor(q0[i], 8, 32);
      q1[i] += __shfl_xor(q1[i], 1, 32);
      q1[i] += __shfl_xor(q1[i], 2, 32);
      q1[i] += __shfl_xor(q1[i], 4, 32);
      q1[i] += __shfl_xor(q1[i], 8, 32);
    }
    if (cn < 8) {
      int i = cn >> 1;                       // sample-in-half 0..3
      int c = cn & 1;                        // output column 0/1
      int sample = (row0 >> 1) + h * 4 + i;  // global sample index
      float val = c ? (q1[i] + bd1) : (q0[i] + bd0);
      out[sample * 2 + c] = val;
    }
  }
}

extern "C" void kernel_launch(void* const* d_in, const int* in_sizes, int n_in,
                              void* d_out, int out_size, void* d_ws, size_t ws_size,
                              hipStream_t stream) {
  (void)n_in; (void)d_ws; (void)ws_size; (void)out_size;
  const float* X  = (const float*)d_in[0];
  // d_in[1] = F_extra, unused by the reference
  const float* WR = (const float*)d_in[2];
  const float* bR = (const float*)d_in[3];
  const float* wA = (const float*)d_in[4];
  const float* WD = (const float*)d_in[5];
  const float* bD = (const float*)d_in[6];
  float* out = (float*)d_out;

  const int B      = in_sizes[0] / (FEAT * 2);  // 65536
  const int nTiles = (2 * B) / 16;              // 8192 wave tiles of 16 rows

  sage_fused_kernel<<<dim3(NUM_WG), dim3(256), LDS_BYTES, stream>>>(
      X, WR, bR, wA, WD, bD, out, nTiles);
}
